// StateSpaceLayer_41996190220672
// MI455X (gfx1250) — matrix operand order repaired and checked by reading, not compiled, not run on hardware
//
#include <hip/hip_runtime.h>

typedef __attribute__((ext_vector_type(16))) _Float16 v16h;
typedef __attribute__((ext_vector_type(8)))  _Float16 v8h;
typedef __attribute__((ext_vector_type(8)))  float    v8f;

#define INPUT_DIM 128
#define STATE_DIM 256
#define OUT_DIM   128
#define NBATCH    32
#define SEQLEN    4096
#define CHUNK     64
#define NCHUNK    (SEQLEN / CHUNK)

#if defined(__HIP_DEVICE_COMPILE__) && __has_builtin(__builtin_amdgcn_global_load_async_to_lds_b128)
#define HAVE_ASYNC_LDS 1
// Exact pointee type per clang diagnostic: int __attribute__((vector_size(16)))
typedef int int4v __attribute__((vector_size(4 * sizeof(int))));
typedef __attribute__((address_space(1))) int4v g_int4;
typedef __attribute__((address_space(3))) int4v l_int4;
#else
#define HAVE_ASYNC_LDS 0
#endif

union V16 { v16h v; v8h h2[2]; };

__device__ __forceinline__ v8f wmma16(v16h a, v16h b, v8f c) {
  // D = A(16x32,f16) x B(32x16,f16) + C(16x16,f32)
  return __builtin_amdgcn_wmma_f32_16x16x32_f16(false, a, false, b, (short)0, c,
                                                false, false);
}

__device__ __forceinline__ void wait_async0() {
#if HAVE_ASYNC_LDS
#if __has_builtin(__builtin_amdgcn_s_wait_asynccnt)
  __builtin_amdgcn_s_wait_asynccnt(0);
#else
  asm volatile("s_wait_asynccnt 0x0" ::: "memory");
#endif
#endif
}

// A-operand fragment from row-major f16 matrix W (leading dim ld, in halves):
// rows 16*tile..16*tile+15, K-slab sl (K = 32*sl..32*sl+31).
// Lane (l,h): elem j -> K = 32sl + 16h + j.
__device__ __forceinline__ v16h load_aop(const _Float16* __restrict__ W, int ld,
                                         int tile, int sl, int l, int h) {
  const _Float16* p = W + (size_t)(tile * 16 + l) * ld + sl * 32 + h * 16;
  V16 u;
  u.h2[0] = *(const v8h*)p;
  u.h2[1] = *(const v8h*)(p + 8);
  return u.v;
}

// B-operand fragment of the shared (transposed) state from the LDS tile dump.
// sbase = &sbuf[par][0][0][0]; tile stride = 32 lanes * 8 halves = 256.
// Lane (l,h), slab sl needs states 32sl+16h..+15 of batch l: dump tile (2sl+h),
// elems 0..7 at sublane (l,h'=0), elems 8..15 at sublane (l,h'=1).
__device__ __forceinline__ v16h read_sfrag(const _Float16* sbase, int sl, int l, int h) {
  const _Float16* p = sbase + (size_t)(sl * 2 + h) * 256 + l * 8;
  V16 u;
  u.h2[0] = *(const v8h*)p;
  u.h2[1] = *(const v8h*)(p + 128);
  return u.v;
}

__device__ __forceinline__ void store8(float* p, v8f a) {
  *(float4*)(p)     = make_float4(a[0], a[1], a[2], a[3]);
  *(float4*)(p + 4) = make_float4(a[4], a[5], a[6], a[7]);
}

__device__ __forceinline__ v8f load8(const float* p) {
  float4 a = *(const float4*)(p);
  float4 b = *(const float4*)(p + 4);
  v8f r;
  r[0] = a.x; r[1] = a.y; r[2] = a.z; r[3] = a.w;
  r[4] = b.x; r[5] = b.y; r[6] = b.z; r[7] = b.w;
  return r;
}

// ---------------------------------------------------------------------------
// f32 -> f16 conversions
// ---------------------------------------------------------------------------
__global__ void cvt_f32_f16_kernel(const float* __restrict__ src,
                                   _Float16* __restrict__ dst, int n) {
  int i = blockIdx.x * blockDim.x + threadIdx.x;
  if (i < n) dst[i] = (_Float16)src[i];
}

// x: exact multiple of 8 elements per thread, bandwidth-bound one-shot pass.
__global__ __launch_bounds__(256)
void cvt_x_kernel(const float* __restrict__ src, _Float16* __restrict__ dst) {
  size_t i = ((size_t)blockIdx.x * blockDim.x + threadIdx.x) * 8;
  float4 a = *(const float4*)(src + i);
  float4 b = *(const float4*)(src + i + 4);
  v8h o;
  o[0] = (_Float16)a.x; o[1] = (_Float16)a.y; o[2] = (_Float16)a.z; o[3] = (_Float16)a.w;
  o[4] = (_Float16)b.x; o[5] = (_Float16)b.y; o[6] = (_Float16)b.z; o[7] = (_Float16)b.w;
  *(v8h*)(dst + i) = o;
}

// ---------------------------------------------------------------------------
// Q = P * P (256x256, f16 in / f32 acc / f16 out). One wave per 16x16 tile.
// Used 6x (log-doubling) to build A^64 for the chunk-combine pass.
// ---------------------------------------------------------------------------
__global__ __launch_bounds__(32, 1)
void matsq_f16_kernel(const _Float16* __restrict__ P, _Float16* __restrict__ Q) {
  const int lane = threadIdx.x & 31, l = lane & 15, h = lane >> 4;
  const int tm = blockIdx.x >> 4, tn = blockIdx.x & 15;
  v8f acc;
#pragma unroll
  for (int v = 0; v < 8; ++v) acc[v] = 0.f;
#pragma unroll
  for (int s = 0; s < 8; ++s) {
    v16h a = load_aop(P, STATE_DIM, tm, s, l, h);
    v16h b;  // B-op[k,n] = P[k, 16tn+l]; elem j -> K = 32s + 16h + j
#pragma unroll
    for (int j = 0; j < 16; ++j)
      b[j] = P[(size_t)(s * 32 + h * 16 + j) * STATE_DIM + tn * 16 + l];
    acc = wmma16(a, b, acc);
  }
#pragma unroll
  for (int v = 0; v < 8; ++v)
    Q[(size_t)(tm * 16 + v + 8 * h) * STATE_DIM + tn * 16 + l] = (_Float16)acc[v];
}

// ---------------------------------------------------------------------------
// Chunk recurrence kernel (pass 1 and pass 3).
// Block = (chunk c, batch-group g of 16). 16 waves; wave w owns state tile w
// (states 16w..16w+15, all 16 batches) as f32 WMMA accumulators.
// Transposed math: s_tT = A*s_{t-1}T + B*x_tT. Weight rows live in VGPRs;
// activations are shared through double-buffered LDS, staged by async copy.
// FINAL also computes y_tT = C*s_tT + D*x_tT on waves 0..7 (one step delayed)
// and dumps final_state for the last chunk.
// ---------------------------------------------------------------------------
template <bool FINAL>
__global__ __launch_bounds__(512, 1)
void ssm_chunk_kernel(const _Float16* __restrict__ xh,
                      const _Float16* __restrict__ wA,
                      const _Float16* __restrict__ wB,
                      const _Float16* __restrict__ wC,
                      const _Float16* __restrict__ wD,
                      const float* __restrict__ carry,   // pass3 in
                      float* __restrict__ dend,          // pass1 out
                      float* __restrict__ y,             // pass3 out
                      float* __restrict__ fstate) {      // pass3 out
  __shared__ __align__(32) _Float16 sbuf[2][16][32][8];   // state frags (16 KB)
  __shared__ __align__(32) _Float16 xbuf[2][4][32][16];   // x_t frags   (8 KB)

  const int tid  = threadIdx.x;
  const int w    = tid >> 5;
  const int lane = tid & 31;
  const int l    = lane & 15;
  const int h    = lane >> 4;
  const int bx   = blockIdx.x;       // = chunk*2 + group
  const int g    = bx & 1;
  const int c    = bx >> 1;
  const int b0   = g * 16;
  const int t0   = c * CHUNK;

  // Loop-invariant weight fragments held in VGPRs for the whole chunk.
  v16h Afr[8], Bfr[4];
#pragma unroll
  for (int s = 0; s < 8; ++s) Afr[s] = load_aop(wA, STATE_DIM, w, s, l, h);
#pragma unroll
  for (int s = 0; s < 4; ++s) Bfr[s] = load_aop(wB, INPUT_DIM, w, s, l, h);
  v16h Cfr[8], Dfr[4];
  if (FINAL && w < 8) {
#pragma unroll
    for (int s = 0; s < 8; ++s) Cfr[s] = load_aop(wC, STATE_DIM, w, s, l, h);
#pragma unroll
    for (int s = 0; s < 4; ++s) Dfr[s] = load_aop(wD, INPUT_DIM, w, s, l, h);
  }

  v8f acc;
  if (FINAL) {
    acc = load8(carry + (((size_t)bx * 16 + w) * 32 + lane) * 8);
  } else {
#pragma unroll
    for (int v = 0; v < 8; ++v) acc[v] = 0.f;
  }
  v8f yacc;
#pragma unroll
  for (int v = 0; v < 8; ++v) yacc[v] = 0.f;

  const int nit = FINAL ? (CHUNK + 1) : CHUNK;  // +1 epilogue iter drains y
  for (int i = 0; i < nit; ++i) {
    const int par = i & 1;
    const int t = t0 + i;

    // ---- stage phase: x_t frags (waves 0..3) + s_{t-1} frag dump (all) ----
    if (w < 4 && i < CHUNK) {
      // B-op layout: lane (l,h) elem j -> K = 32w + 16h + j (32B contiguous f16)
      const _Float16* xg = xh + ((size_t)(b0 + l) * SEQLEN + t) * INPUT_DIM
                              + w * 32 + h * 16;
#if HAVE_ASYNC_LDS
      __builtin_amdgcn_global_load_async_to_lds_b128(
          (g_int4*)xg, (l_int4*)&xbuf[par][w][lane][0], 0, 0);
      __builtin_amdgcn_global_load_async_to_lds_b128(
          (g_int4*)(xg + 8), (l_int4*)&xbuf[par][w][lane][8], 0, 0);
#else
      V16 u;
      u.h2[0] = *(const v8h*)xg;
      u.h2[1] = *(const v8h*)(xg + 8);
      *(v16h*)&xbuf[par][w][lane][0] = u.v;
#endif
      int tp = t + 16;
      if (tp > SEQLEN - 1) tp = SEQLEN - 1;
      __builtin_prefetch(xh + ((size_t)(b0 + l) * SEQLEN + tp) * INPUT_DIM
                            + w * 32 + h * 16, 0, 1);
      wait_async0();
    }
    {
      v8h sv;
#pragma unroll
      for (int v = 0; v < 8; ++v) sv[v] = (_Float16)acc[v];
      *(v8h*)&sbuf[par][w][lane][0] = sv;
    }
    __syncthreads();
    const _Float16* sbase = &sbuf[par][0][0][0];

    // ---- finish y_{t-1}: yacc holds D*x_{t-1}; add C*s_{t-1} (2 chains) ----
    if (FINAL && w < 8 && i > 0) {
      v8f ya0 = yacc, ya1;
#pragma unroll
      for (int v = 0; v < 8; ++v) ya1[v] = 0.f;
#pragma unroll
      for (int s = 0; s < 4; ++s) {
        ya0 = wmma16(Cfr[s],     read_sfrag(sbase, s,     l, h), ya0);
        ya1 = wmma16(Cfr[s + 4], read_sfrag(sbase, s + 4, l, h), ya1);
      }
      ya0 = ya0 + ya1;
      store8(y + ((size_t)(b0 + l) * SEQLEN + (t - 1)) * OUT_DIM + w * 16 + h * 8, ya0);
    }

    // ---- state update: s_t = B*x_t + A*s_{t-1} (two 6-deep WMMA chains) ----
    if (i < CHUNK) {
      v16h xf[4];
#pragma unroll
      for (int s = 0; s < 4; ++s) xf[s] = *(const v16h*)&xbuf[par][s][lane][0];
      v8f na0, na1;
#pragma unroll
      for (int v = 0; v < 8; ++v) { na0[v] = 0.f; na1[v] = 0.f; }
      na0 = wmma16(Bfr[0], xf[0], na0);
      na1 = wmma16(Bfr[1], xf[1], na1);
      na0 = wmma16(Bfr[2], xf[2], na0);
      na1 = wmma16(Bfr[3], xf[3], na1);
#pragma unroll
      for (int s = 0; s < 4; ++s) {
        na0 = wmma16(Afr[s],     read_sfrag(sbase, s,     l, h), na0);
        na1 = wmma16(Afr[s + 4], read_sfrag(sbase, s + 4, l, h), na1);
      }
      acc = na0 + na1;
      if (FINAL && w < 8) {
        v8f d0, d1;
#pragma unroll
        for (int v = 0; v < 8; ++v) { d0[v] = 0.f; d1[v] = 0.f; }
        d0 = wmma16(Dfr[0], xf[0], d0);
        d1 = wmma16(Dfr[1], xf[1], d1);
        d0 = wmma16(Dfr[2], xf[2], d0);
        d1 = wmma16(Dfr[3], xf[3], d1);
        yacc = d0 + d1;
      }
    }
  }

  if (!FINAL) {
    store8(dend + (((size_t)bx * 16 + w) * 32 + lane) * 8, acc);
  } else if (c == NCHUNK - 1) {
    store8(fstate + (size_t)(b0 + l) * STATE_DIM + w * 16 + h * 8, acc);
  }
}

// ---------------------------------------------------------------------------
// Pass 2: sequential combine over chunks.
// carry[c] = state entering chunk c; carry[c] = carry[c-1]*(A^L)^T + dend[c-1].
// Transposed: carryT_new = G*carryT + dendT, G = A^64 (f16 rows in VGPRs).
// ---------------------------------------------------------------------------
__global__ __launch_bounds__(512, 1)
void ssm_combine_kernel(const _Float16* __restrict__ G,
                        const float* __restrict__ dend,
                        float* __restrict__ carry) {
  __shared__ __align__(32) _Float16 sbuf[2][16][32][8];
  const int tid = threadIdx.x;
  const int w = tid >> 5, lane = tid & 31, l = lane & 15, h = lane >> 4;
  const int g = blockIdx.x;

  v16h Gfr[8];
#pragma unroll
  for (int s = 0; s < 8; ++s) Gfr[s] = load_aop(G, STATE_DIM, w, s, l, h);

  v8f acc;
#pragma unroll
  for (int v = 0; v < 8; ++v) acc[v] = 0.f;

  for (int c = 0; c < NCHUNK; ++c) {
    const int par = c & 1;
    const size_t idx = (((size_t)(c * 2 + g) * 16 + w) * 32 + lane) * 8;
    store8(carry + idx, acc);
    v8h sv;
#pragma unroll
    for (int v = 0; v < 8; ++v) sv[v] = (_Float16)acc[v];
    *(v8h*)&sbuf[par][w][lane][0] = sv;
    __syncthreads();
    const _Float16* sbase = &sbuf[par][0][0][0];
    v8f na0 = load8(dend + idx), na1;
#pragma unroll
    for (int v = 0; v < 8; ++v) na1[v] = 0.f;
#pragma unroll
    for (int s = 0; s < 4; ++s) {
      na0 = wmma16(Gfr[s],     read_sfrag(sbase, s,     l, h), na0);
      na1 = wmma16(Gfr[s + 4], read_sfrag(sbase, s + 4, l, h), na1);
    }
    acc = na0 + na1;
  }
}

// ---------------------------------------------------------------------------
extern "C" void kernel_launch(void* const* d_in, const int* in_sizes, int n_in,
                              void* d_out, int out_size, void* d_ws, size_t ws_size,
                              hipStream_t stream) {
  (void)in_sizes; (void)n_in; (void)out_size; (void)ws_size;
  const float* x  = (const float*)d_in[0];
  const float* A  = (const float*)d_in[1];
  const float* B  = (const float*)d_in[2];
  const float* Cm = (const float*)d_in[3];
  const float* D  = (const float*)d_in[4];
  float* y = (float*)d_out;
  float* fstate = y + (size_t)NBATCH * SEQLEN * OUT_DIM;

  const size_t NX = (size_t)NBATCH * SEQLEN * INPUT_DIM;  // 16,777,216
  char* ws = (char*)d_ws;
  _Float16* xh = (_Float16*)ws;  ws += NX * 2;                            // 32 MB
  _Float16* wA = (_Float16*)ws;  ws += (size_t)STATE_DIM * STATE_DIM * 2;
  _Float16* wB = (_Float16*)ws;  ws += (size_t)STATE_DIM * INPUT_DIM * 2;
  _Float16* wC = (_Float16*)ws;  ws += (size_t)OUT_DIM * STATE_DIM * 2;
  _Float16* wD = (_Float16*)ws;  ws += (size_t)OUT_DIM * INPUT_DIM * 2;
  _Float16* Pa = (_Float16*)ws;  ws += (size_t)STATE_DIM * STATE_DIM * 2;
  _Float16* Pb = (_Float16*)ws;  ws += (size_t)STATE_DIM * STATE_DIM * 2;
  float* dend  = (float*)ws;     ws += (size_t)NCHUNK * 2 * 16 * 32 * 8 * 4;
  float* carry = (float*)ws;

  // One-shot conversions (bandwidth-bound, fully parallel).
  cvt_x_kernel<<<(int)(NX / 8 / 256), 256, 0, stream>>>(x, xh);
  cvt_f32_f16_kernel<<<(65536 + 255) / 256, 256, 0, stream>>>(A,  wA, 65536);
  cvt_f32_f16_kernel<<<(32768 + 255) / 256, 256, 0, stream>>>(B,  wB, 32768);
  cvt_f32_f16_kernel<<<(32768 + 255) / 256, 256, 0, stream>>>(Cm, wC, 32768);
  cvt_f32_f16_kernel<<<(16384 + 255) / 256, 256, 0, stream>>>(D,  wD, 16384);

  // G = A^64 via 6 log-doubling squarings.
  matsq_f16_kernel<<<256, 32, 0, stream>>>(wA, Pa);  // A^2
  matsq_f16_kernel<<<256, 32, 0, stream>>>(Pa, Pb);  // A^4
  matsq_f16_kernel<<<256, 32, 0, stream>>>(Pb, Pa);  // A^8
  matsq_f16_kernel<<<256, 32, 0, stream>>>(Pa, Pb);  // A^16
  matsq_f16_kernel<<<256, 32, 0, stream>>>(Pb, Pa);  // A^32
  matsq_f16_kernel<<<256, 32, 0, stream>>>(Pa, Pb);  // A^64

  // Pass 1: per-chunk local scan (zero init) -> chunk-end states.
  ssm_chunk_kernel<false><<<NCHUNK * 2, 512, 0, stream>>>(
      xh, wA, wB, wC, wD, nullptr, dend, nullptr, nullptr);
  // Pass 2: sequential carry combine across chunks.
  ssm_combine_kernel<<<2, 512, 0, stream>>>(Pb, dend, carry);
  // Pass 3: seeded per-chunk scan fused with y = C*s + D*x, + final_state.
  ssm_chunk_kernel<true><<<NCHUNK * 2, 512, 0, stream>>>(
      xh, wA, wB, wC, wD, carry, nullptr, y, fstate);
}
